// TreeModel_34359738368103
// MI455X (gfx1250) — matrix operand order repaired and checked by reading, not compiled, not run on hardware
//
#include <hip/hip_runtime.h>
#include <hip/hip_bf16.h>
#include <math.h>

// ---------------------------------------------------------------------------
// TreeLSTM over complete binary tree, depth 18 (N = 524287 nodes).
//   * bf16 WMMA (v_wmma_f32_16x16x32_bf16) for all matmuls, fp32 accumulate.
//   * Pre-pack:  emb -> bf16 [50000 x 320] (K padded 300->320; 32MB -> L2)
//                Wt  -> bf16 [608 x 320]  transposed [W_iou | W_f | pad]
//                Ut  -> bf16 [608 x 160]  transposed [U_iou | U_f | pad]
//   * Per level: one workgroup (8 waves) per 16-node M-tile computes
//                xw[16][608] = emb_rows @ [W_iou|W_f]      (K=320)
//                g [32][608] = child h rows @ [U_iou|U_f]  (K=160)
//                in LDS (row stride 612 to kill store bank conflicts),
//                then fused elementwise LSTM cell -> h,c (fp32, ld=160).
//   * 19 sequential level launches on `stream` give the tree dependency.
//   * A fragments hoisted to VGPRs; B chunks software-pipelined; Ut B-frag
//     shared by two independent child accumulators (2 WMMAs per load).
// Workspace: emb_bf 32MB + Wt 380KB + Ut 190KB + h 336MB + c 336MB ~= 704MB.
// ---------------------------------------------------------------------------

typedef __attribute__((ext_vector_type(8)))  __bf16 v8bf;
typedef __attribute__((ext_vector_type(16))) __bf16 v16bf;
typedef __attribute__((ext_vector_type(8)))  float  v8f;

#define VOCAB   50000
#define XS      300
#define XSP     320     // padded K for x-GEMM (10 chunks of 32)
#define HS      150
#define HSP     160     // padded K for h-GEMM (5 chunks of 32)
#define NCOL    608     // padded combined N (38 tiles of 16): [iou 450 | f 150 | pad]
#define LDSW    612     // LDS row stride (floats): 8*612 % 64 == 32 -> no bank conflict
#define NNODES  524287

static const size_t WT_OFF = (size_t)VOCAB * XSP * 2;           // 32,000,000
static const size_t UT_OFF = WT_OFF + (size_t)NCOL * XSP * 2;   // +389,120
static const size_t H_OFF  = UT_OFF + (size_t)NCOL * HSP * 2;   // +194,560
static const size_t C_OFF  = H_OFF + (size_t)NNODES * HSP * 4;  // +335,543,680

__device__ __forceinline__ float sigmoidf_(float x) { return 1.0f / (1.0f + __expf(-x)); }

// A-fragment (16x32 bf16): lane holds row lane&15; contiguous K run of 16
// at koff = half*16  (half = lane>>4).
__device__ __forceinline__ v16bf frag_a_bf16(const __bf16* __restrict__ row, int half) {
  const __bf16* p = row + half * 16;
  v8bf c0 = *(const v8bf*)(p);
  v8bf c1 = *(const v8bf*)(p + 8);
  v16bf a;
#pragma unroll
  for (int e = 0; e < 8; ++e) { a[e] = c0[e]; a[e + 8] = c1[e]; }
  return a;
}

// Same A layout, source is fp32 (h rows) converted in-register to bf16.
__device__ __forceinline__ v16bf frag_a_f32(const float* __restrict__ row, int half) {
  const float* p = row + half * 16;
  v16bf a;
#pragma unroll
  for (int e = 0; e < 8; ++e) { a[e] = (__bf16)p[e]; a[e + 8] = (__bf16)p[e + 8]; }
  return a;
}

// B-fragment (32x16 bf16): B stored transposed (row = output col n, col = k).
// Lane holds column lane&15, contiguous K run of 16 at (lane>>4)*16 (folded
// into the pointer by the caller).
__device__ __forceinline__ v16bf frag_b(const __bf16* __restrict__ p) {
  v8bf c0 = *(const v8bf*)(p);
  v8bf c1 = *(const v8bf*)(p + 8);
  v16bf b;
#pragma unroll
  for (int e = 0; e < 8; ++e) { b[e] = c0[e]; b[e + 8] = c1[e]; }
  return b;
}

// ------------------------------ pack kernels -------------------------------

__global__ void pack_emb_kernel(const float* __restrict__ emb, __bf16* __restrict__ emb_bf) {
  int row = blockIdx.x;
  int t = threadIdx.x; // 0..319
  float v = (t < XS) ? emb[(size_t)row * XS + t] : 0.0f;
  emb_bf[(size_t)row * XSP + t] = (__bf16)v;
}

__global__ void pack_w_kernel(const float* __restrict__ W_iou, const float* __restrict__ W_f,
                              __bf16* __restrict__ Wt) {
  int n = blockIdx.x;  // 0..607 (output column)
  int k = threadIdx.x; // 0..319
  float v = 0.0f;
  if (k < XS) {
    if (n < 450)      v = W_iou[(size_t)k * 450 + n];
    else if (n < 600) v = W_f[(size_t)k * 150 + (n - 450)];
  }
  Wt[(size_t)n * XSP + k] = (__bf16)v;
}

__global__ void pack_u_kernel(const float* __restrict__ U_iou, const float* __restrict__ U_f,
                              __bf16* __restrict__ Ut) {
  int n = blockIdx.x;  // 0..607
  int k = threadIdx.x; // 0..159
  float v = 0.0f;
  if (k < HS) {
    if (n < 450)      v = U_iou[(size_t)k * 450 + n];
    else if (n < 600) v = U_f[(size_t)k * 150 + (n - 450)];
  }
  Ut[(size_t)n * HSP + k] = (__bf16)v;
}

// ------------------------------ level kernel -------------------------------
// One workgroup (256 threads = 8 waves) per 16-node M-tile of the level.
// Wave w handles N-tiles {w, w+8, ...} of the 608-wide combined output.

__global__ __launch_bounds__(256) void level_kernel(
    const int* __restrict__ x_id, const __bf16* __restrict__ emb_bf,
    const __bf16* __restrict__ Wt, const __bf16* __restrict__ Ut,
    const float* __restrict__ b_iou, const float* __restrict__ b_f,
    float* __restrict__ h, float* __restrict__ c,
    int s, int count, int has_children) {
  __shared__ float xw[16 * LDSW]; // x @ [W_iou|W_f]
  __shared__ float g[32 * LDSW];  // h_child @ [U_iou|U_f]

  const int tid  = threadIdx.x;
  const int lane = tid & 31;
  const int wave = tid >> 5;
  const int half = lane >> 4;
  const int col  = lane & 15;
  const int mbase = blockIdx.x * 16;

  // ---- hoist all A fragments into VGPRs (reused across every N-tile) ----
  int rowm = mbase + col;
  if (rowm >= count) rowm = 0;
  const __bf16* arow_x = emb_bf + (size_t)x_id[s + rowm] * XSP;
  v16bf ax[XSP / 32];
#pragma unroll
  for (int kk = 0; kk < XSP / 32; ++kk) ax[kk] = frag_a_bf16(arow_x + kk * 32, half);

  v16bf ag0[HSP / 32], ag1[HSP / 32];
  if (has_children) {
    int chbase = 2 * (s + mbase) + 1;
    int maxr = 2 * (count - mbase);
    int r0 = col;      if (r0 >= maxr) r0 = 0;
    int r1 = 16 + col; if (r1 >= maxr) r1 = 0;
    const float* ah0 = h + (size_t)(chbase + r0) * HSP;
    const float* ah1 = h + (size_t)(chbase + r1) * HSP;
#pragma unroll
    for (int kk = 0; kk < HSP / 32; ++kk) {
      ag0[kk] = frag_a_f32(ah0 + kk * 32, half);
      ag1[kk] = frag_a_f32(ah1 + kk * 32, half);
    }
  }

  if (has_children) {
    for (int nt = wave; nt < NCOL / 16; nt += 8) {
      const int scol = nt * 16 + col;
      // ---- xw tile: K = 320, software-pipelined B ----
      const __bf16* wrow = Wt + (size_t)scol * XSP + half * 16;
      v16bf b = frag_b(wrow);
      v8f acc = {};
#pragma unroll
      for (int kk = 0; kk < XSP / 32; ++kk) {
        v16bf bn = b;
        if (kk + 1 < XSP / 32) bn = frag_b(wrow + (kk + 1) * 32);
        acc = __builtin_amdgcn_wmma_f32_16x16x32_bf16(false, ax[kk], false, b, (short)0, acc, false, false);
        b = bn;
      }
#pragma unroll
      for (int j = 0; j < 8; ++j) xw[(half * 8 + j) * LDSW + scol] = acc[j];

      // ---- g tiles: one Ut B-frag feeds two independent accumulators ----
      const __bf16* urow = Ut + (size_t)scol * HSP + half * 16;
      v16bf bu = frag_b(urow);
      v8f acc0 = {}, acc1 = {};
#pragma unroll
      for (int kk = 0; kk < HSP / 32; ++kk) {
        v16bf bun = bu;
        if (kk + 1 < HSP / 32) bun = frag_b(urow + (kk + 1) * 32);
        acc0 = __builtin_amdgcn_wmma_f32_16x16x32_bf16(false, ag0[kk], false, bu, (short)0, acc0, false, false);
        acc1 = __builtin_amdgcn_wmma_f32_16x16x32_bf16(false, ag1[kk], false, bu, (short)0, acc1, false, false);
        bu = bun;
      }
#pragma unroll
      for (int j = 0; j < 8; ++j) {
        g[(half * 8 + j) * LDSW + scol]        = acc0[j];
        g[(16 + half * 8 + j) * LDSW + scol]   = acc1[j];
      }
    }
  } else {
    for (int nt = wave; nt < 29; nt += 8) { // leaves: iou columns only
      const int scol = nt * 16 + col;
      const __bf16* wrow = Wt + (size_t)scol * XSP + half * 16;
      v16bf b = frag_b(wrow);
      v8f acc = {};
#pragma unroll
      for (int kk = 0; kk < XSP / 32; ++kk) {
        v16bf bn = b;
        if (kk + 1 < XSP / 32) bn = frag_b(wrow + (kk + 1) * 32);
        acc = __builtin_amdgcn_wmma_f32_16x16x32_bf16(false, ax[kk], false, b, (short)0, acc, false, false);
        b = bn;
      }
#pragma unroll
      for (int j = 0; j < 8; ++j) xw[(half * 8 + j) * LDSW + scol] = acc[j];
    }
  }
  __syncthreads();

  // ---- fused elementwise LSTM cell ----
  for (int idx = tid; idx < 16 * HSP; idx += 256) {
    int i = idx / HSP;
    int d = idx - i * HSP;
    int m = mbase + i;
    if (m >= count) continue;
    int n = s + m;
    float hval = 0.0f, cval = 0.0f; // cols 150..159 stay zero (K padding of h/c)
    if (d < HS) {
      float iv = xw[i * LDSW + d]          + b_iou[d];
      float ov = xw[i * LDSW + HS + d]     + b_iou[HS + d];
      float uv = xw[i * LDSW + 2 * HS + d] + b_iou[2 * HS + d];
      float fc = 0.0f;
      if (has_children) {
        const float* gl = g + (size_t)(2 * i) * LDSW;
        const float* gr = g + (size_t)(2 * i + 1) * LDSW;
        iv += gl[d] + gr[d];
        ov += gl[HS + d] + gr[HS + d];
        uv += gl[2 * HS + d] + gr[2 * HS + d];
        float xf = xw[i * LDSW + 3 * HS + d] + b_f[d];
        float fl = sigmoidf_(xf + gl[3 * HS + d]);
        float fr = sigmoidf_(xf + gr[3 * HS + d]);
        fc = fl * c[(size_t)(2 * n + 1) * HSP + d] + fr * c[(size_t)(2 * n + 2) * HSP + d];
      }
      cval = sigmoidf_(iv) * tanhf(uv) + fc;
      hval = sigmoidf_(ov) * tanhf(cval);
    }
    h[(size_t)n * HSP + d] = hval;
    c[(size_t)n * HSP + d] = cval;
  }
}

// ------------------------------ classifier head ----------------------------

__global__ __launch_bounds__(256) void out_kernel(const float* __restrict__ h,
                                                  const float* __restrict__ W_out,
                                                  const float* __restrict__ b_out,
                                                  float* __restrict__ out, int n_nodes) {
  int n = blockIdx.x * blockDim.x + threadIdx.x;
  if (n >= n_nodes) return;
  float a0 = b_out[0], a1 = b_out[1], a2 = b_out[2], a3 = b_out[3], a4 = b_out[4];
  const float* hr = h + (size_t)n * HSP;
  for (int d = 0; d < HS; ++d) {
    float hv = hr[d];
    a0 += hv * W_out[d * 5 + 0];
    a1 += hv * W_out[d * 5 + 1];
    a2 += hv * W_out[d * 5 + 2];
    a3 += hv * W_out[d * 5 + 3];
    a4 += hv * W_out[d * 5 + 4];
  }
  float* o = out + (size_t)n * 5;
  o[0] = a0; o[1] = a1; o[2] = a2; o[3] = a3; o[4] = a4;
}

// ------------------------------ launch -------------------------------------

extern "C" void kernel_launch(void* const* d_in, const int* in_sizes, int n_in,
                              void* d_out, int out_size, void* d_ws, size_t ws_size,
                              hipStream_t stream) {
  const int*   x_id  = (const int*)  d_in[0];
  const float* emb   = (const float*)d_in[1];
  const float* W_iou = (const float*)d_in[2];
  const float* U_iou = (const float*)d_in[3];
  const float* b_iou = (const float*)d_in[4];
  const float* W_f   = (const float*)d_in[5];
  const float* U_f   = (const float*)d_in[6];
  const float* b_f   = (const float*)d_in[7];
  const float* W_out = (const float*)d_in[8];
  const float* b_out = (const float*)d_in[9];

  char* ws = (char*)d_ws;
  __bf16* emb_bf = (__bf16*)(ws);
  __bf16* Wt     = (__bf16*)(ws + WT_OFF);
  __bf16* Ut     = (__bf16*)(ws + UT_OFF);
  float*  h      = (float*) (ws + H_OFF);
  float*  c      = (float*) (ws + C_OFF);

  pack_emb_kernel<<<VOCAB, XSP, 0, stream>>>(emb, emb_bf);
  pack_w_kernel<<<NCOL, XSP, 0, stream>>>(W_iou, W_f, Wt);
  pack_u_kernel<<<NCOL, HSP, 0, stream>>>(U_iou, U_f, Ut);

  const int depth = 18;
  for (int lvl = depth; lvl >= 0; --lvl) {
    int s = (1 << lvl) - 1;
    int count = 1 << lvl;
    int blocks = (count + 15) / 16;
    level_kernel<<<blocks, 256, 0, stream>>>(x_id, emb_bf, Wt, Ut, b_iou, b_f,
                                             h, c, s, count, (lvl != depth) ? 1 : 0);
  }

  int n_nodes = in_sizes[0];
  out_kernel<<<(n_nodes + 255) / 256, 256, 0, stream>>>(h, W_out, b_out, (float*)d_out, n_nodes);
}
